// MultiHeadAttention_32667521254080
// MI455X (gfx1250) — matrix operand order repaired and checked by reading, not compiled, not run on hardware
//
#include <hip/hip_runtime.h>
#include <hip/hip_bf16.h>
#include <stdint.h>

// ---------------------------------------------------------------------------
// MI455X (gfx1250) fused GAT-style attention.
//   out = softmax(u 1^T + 1 v^T, axis=1) @ op1_flat
// Never materializes the 8192x8192 score matrix (512 MB of HBM traffic saved);
// weights are generated in-register per 32-K chunk and fed to
// v_wmma_f32_16x16x32_f16 (f16 inputs, f32 accumulate).
// ---------------------------------------------------------------------------

typedef _Float16     v16h __attribute__((ext_vector_type(16)));
typedef float        v8f  __attribute__((ext_vector_type(8)));
typedef __fp16       h2   __attribute__((ext_vector_type(2)));  // cvt_pkrtz result type
typedef float        f4   __attribute__((ext_vector_type(4)));
typedef unsigned int u4   __attribute__((ext_vector_type(4)));

#define NN  8192
#define DD  16
#define DHH 18
#define HH  8
#define HD  128   // H*D

union X16    { f4 f[4];  float s[16]; };
union Frag16 { v16h v;   h2 p[8];  u4 q[2]; };
union Acc8   { v8f v;    f4 f[2];  float s[8]; };

// ---------------------------------------------------------------------------
// Kernel 1: per row j compute
//   u[j] = inputs[j,:] . (W @ a1)      v[j] = inputs[j,:] . (W @ a2)
//   op1T[h*16+e][j] = (f16) sum_k inputs[j,k] * Wk[h,k,e]   (transposed f16)
// Wk (8 KB) staged in LDS. One thread per row j.
// ---------------------------------------------------------------------------
__global__ __launch_bounds__(256) void prep_kernel(
    const float* __restrict__ inputs, const float* __restrict__ W,
    const float* __restrict__ avec,   const float* __restrict__ Wk,
    float* __restrict__ u, float* __restrict__ v, _Float16* __restrict__ op1T)
{
    __shared__ float sW1[DD];
    __shared__ float sW2[DD];
    __shared__ float sWk[HH * DD * DD];

    const int t = threadIdx.x;
    for (int idx = t; idx < HH * DD * DD; idx += 256) sWk[idx] = Wk[idx];
    if (t < 32) {
        const int k = t & 15;
        const float* ap = avec + ((t < 16) ? 0 : DHH);
        float s = 0.0f;
        #pragma unroll
        for (int q = 0; q < DHH; ++q) s += W[k * DHH + q] * ap[q];
        if (t < 16) sW1[k] = s; else sW2[k] = s;
    }
    __syncthreads();

    const int j = blockIdx.x * 256 + t;
    X16 X;
    const f4* xp = (const f4*)(inputs + (size_t)j * DD);
    X.f[0] = xp[0]; X.f[1] = xp[1]; X.f[2] = xp[2]; X.f[3] = xp[3];

    float du = 0.0f, dv = 0.0f;
    #pragma unroll
    for (int k = 0; k < DD; ++k) { du += X.s[k] * sW1[k]; dv += X.s[k] * sW2[k]; }
    u[j] = du; v[j] = dv;

    const f4 zf = {0.f, 0.f, 0.f, 0.f};
    for (int h = 0; h < HH; ++h) {
        X16 acc;
        #pragma unroll
        for (int d4 = 0; d4 < 4; ++d4) acc.f[d4] = zf;
        #pragma unroll
        for (int k = 0; k < DD; ++k) {
            const f4* wr = (const f4*)(sWk + (h * DD + k) * DD);
            const float xk = X.s[k];
            acc.f[0] += xk * wr[0];
            acc.f[1] += xk * wr[1];
            acc.f[2] += xk * wr[2];
            acc.f[3] += xk * wr[3];
        }
        _Float16* orow = op1T + (size_t)(h * DD) * NN + j;   // coalesced over j
        #pragma unroll
        for (int d = 0; d < DD; ++d) orow[(size_t)d * NN] = (_Float16)acc.s[d];
    }
}

// ---------------------------------------------------------------------------
// Kernel 2: maxv = max_j v[j]   (single block tree reduction over 8192 values)
// ---------------------------------------------------------------------------
__global__ __launch_bounds__(256) void maxred_kernel(
    const float* __restrict__ v, float* __restrict__ maxv)
{
    __shared__ float red[256];
    const int t = threadIdx.x;
    float m = -3.402823466e38f;
    for (int j = t; j < NN; j += 256) m = fmaxf(m, v[j]);
    red[t] = m;
    __syncthreads();
    for (int s = 128; s > 0; s >>= 1) {
        if (t < s) red[t] = fmaxf(red[t], red[t + s]);
        __syncthreads();
    }
    if (t == 0) *maxv = red[0];
}

// ---------------------------------------------------------------------------
// Kernel 3: fused softmax x GEMM via WMMA, computing out^T = op1^T . score^T.
// Each wave owns 32 output rows (two 16-row i-tiles, lane%16 = i).
//   A frag  : op1T tile (M=c,K=j) -> two contiguous 16B loads per lane
//   B frag  : w[j,i] = exp2(v[j]*log2e + (u[i]-rowmax[i])*log2e), one
//             v_fma + v_exp per weight, packed with v_cvt_pk_rtz_f16_f32
//   D frag  : lane holds 8 consecutive output columns of row i -> b128 stores
// rowsum accumulated per-lane alongside; normalization is a per-lane scalar.
// A periodic block barrier keeps the 8 waves convergent so their shared
// op1T stream stays resident in the WGP$ instead of replaying from L2.
// ---------------------------------------------------------------------------
__global__ __launch_bounds__(256) void attn_kernel(
    const float* __restrict__ u, const float* __restrict__ v,
    const float* __restrict__ maxvp, const _Float16* __restrict__ op1T,
    float* __restrict__ out)
{
    const int lane = threadIdx.x & 31;
    const int wv   = threadIdx.x >> 5;
    const int li   = lane & 15;
    const int hi   = lane >> 4;
    const int gw   = blockIdx.x * 8 + wv;
    const int ia   = gw * 32 + li;
    const int ib   = ia + 16;

    const float LOG2E = 1.4426950408889634f;
    const float maxv = *maxvp;
    const float ua = u[ia], ub = u[ib];
    const float rma = ua + maxv, rmb = ub + maxv;   // per-row softmax max
    const float caa = (ua - rma) * LOG2E;           // per-lane exp2 bias
    const float cab = (ub - rmb) * LOG2E;

    v8f Ca[8], Cb[8];
    const v8f vz = {0.f, 0.f, 0.f, 0.f, 0.f, 0.f, 0.f, 0.f};
    #pragma unroll
    for (int tt = 0; tt < 8; ++tt) { Ca[tt] = vz; Cb[tt] = vz; }
    float rsa = 0.0f, rsb = 0.0f;

    const _Float16* pA = op1T + (size_t)li * NN + 16 * hi; // A-frag lane base
    const float*    pV = v + 16 * hi;                      // B-frag K = e+16*hi

    for (int jb = 0; jb < NN; jb += 32) {
        if ((jb & 255) == 0) __syncthreads();   // keep waves converged on op1T stream

        X16 V;
        const f4* vp = (const f4*)(pV + jb);
        V.f[0] = vp[0]; V.f[1] = vp[1]; V.f[2] = vp[2]; V.f[3] = vp[3];

        // Build both B fragments (softmax weights for the two i-tiles).
        Frag16 Ba, Bb;
        #pragma unroll
        for (int e = 0; e < 16; e += 2) {
            const float wa0 = __builtin_amdgcn_exp2f(fmaf(V.s[e],     LOG2E, caa));
            const float wa1 = __builtin_amdgcn_exp2f(fmaf(V.s[e + 1], LOG2E, caa));
            const float wb0 = __builtin_amdgcn_exp2f(fmaf(V.s[e],     LOG2E, cab));
            const float wb1 = __builtin_amdgcn_exp2f(fmaf(V.s[e + 1], LOG2E, cab));
            rsa += wa0 + wa1;
            rsb += wb0 + wb1;
            Ba.p[e >> 1] = __builtin_amdgcn_cvt_pkrtz(wa0, wa1);
            Bb.p[e >> 1] = __builtin_amdgcn_cvt_pkrtz(wb0, wb1);
        }

        // 8 c-tiles x 2 i-tiles = 16 WMMAs; A fragment shared between them.
        #pragma unroll
        for (int tt = 0; tt < 8; ++tt) {
            Frag16 A;
            const u4* ap = (const u4*)(pA + (size_t)tt * (16 * NN) + jb);
            A.q[0] = ap[0];   // K = jb + 16*hi     .. +7
            A.q[1] = ap[1];   // K = jb + 16*hi + 8 .. +15
            Ca[tt] = __builtin_amdgcn_wmma_f32_16x16x32_f16(
                false, A.v, false, Ba.v, (short)0, Ca[tt], false, false);
            Cb[tt] = __builtin_amdgcn_wmma_f32_16x16x32_f16(
                false, A.v, false, Bb.v, (short)0, Cb[tt], false, false);
        }
    }

    // Each lane summed half the K range for its row; fold across hi halves.
    rsa += __shfl_xor(rsa, 16, 32);
    rsb += __shfl_xor(rsb, 16, 32);
    const float inva = 1.0f / rsa;
    const float invb = 1.0f / rsb;

    float* rowa = out + (size_t)ia * HD + 8 * hi;
    float* rowb = out + (size_t)ib * HD + 8 * hi;
    #pragma unroll
    for (int tt = 0; tt < 8; ++tt) {
        Acc8 da; da.v = Ca[tt] * inva;
        *(f4*)(rowa + tt * 16)     = da.f[0];
        *(f4*)(rowa + tt * 16 + 4) = da.f[1];
        Acc8 db; db.v = Cb[tt] * invb;
        *(f4*)(rowb + tt * 16)     = db.f[0];
        *(f4*)(rowb + tt * 16 + 4) = db.f[1];
    }
}

// ---------------------------------------------------------------------------
// Launcher. Workspace layout:
//   [0, 32768)          u (8192 f32)
//   [32768, 65536)      v (8192 f32)
//   [65536, 65792)      maxv scalar (padded)
//   [65792, 65792+2MB)  op1T (128 x 8192 f16)
// ---------------------------------------------------------------------------
extern "C" void kernel_launch(void* const* d_in, const int* in_sizes, int n_in,
                              void* d_out, int out_size, void* d_ws, size_t ws_size,
                              hipStream_t stream) {
    const float* inputs = (const float*)d_in[0];
    const float* W      = (const float*)d_in[1];
    const float* avec   = (const float*)d_in[2];
    const float* Wk     = (const float*)d_in[3];
    float* out = (float*)d_out;

    char* ws = (char*)d_ws;
    float*    u    = (float*)(ws);
    float*    v    = (float*)(ws + 32768);
    float*    maxv = (float*)(ws + 65536);
    _Float16* op1T = (_Float16*)(ws + 65792);

    prep_kernel  <<<NN / 256, 256, 0, stream>>>(inputs, W, avec, Wk, u, v, op1T);
    maxred_kernel<<<1,        256, 0, stream>>>(v, maxv);
    attn_kernel  <<<NN / 256, 256, 0, stream>>>(u, v, maxv, op1T, out); // 32 blocks x 8 waves x 32 rows
}
